// SelfAttention_42700564857220
// MI455X (gfx1250) — compile-verified, matrix operand order checked by reading, not run on hardware
//
#include <hip/hip_runtime.h>

// ---------------------------------------------------------------------------
// MI455X (gfx1250) fused self-attention forward.
// - All GEMM work on v_wmma_f32_16x16x32_bf16 (f32 accumulate).
// - GEMM A-tiles staged via GLOBAL_LOAD_ASYNC_TO_LDS_B128 with double
//   buffering (ASYNCcnt-tracked DMA overlapped with WMMA compute).
// ---------------------------------------------------------------------------

typedef __bf16 bf16_t;
typedef __attribute__((ext_vector_type(16))) __bf16 v16bf;
typedef __attribute__((ext_vector_type(8)))  float  v8f;

static constexpr int Bc  = 4;
static constexpr int Tc  = 2048;
static constexpr int Dc  = 1024;
static constexpr int Hc  = 16;
static constexpr int DKc = 64;
static constexpr int BTc = Bc * Tc;

// ISA 16-bit A/B fragment K index for a given fragment slot (0..15).
// lane<16 holds K = 0..7 and 16..23 ; lane>=16 holds K = 8..15 and 24..31.
__device__ __forceinline__ int kmap(int idx, int half) {
    return (idx < 8 ? idx : idx + 8) + 8 * half;
}

// Async global->LDS copy, 16B per lane (VGLOBAL GV mode, ASYNCcnt-tracked).
// lds_byte: LDS byte address (low 32 bits of the generic shared pointer;
// flat->LDS mapping discards the upper bits per the ISA aperture rules).
__device__ __forceinline__ void async_copy_b128(unsigned lds_byte,
                                                unsigned long long gaddr) {
    asm volatile("global_load_async_to_lds_b128 %0, %1, off"
                 :: "v"(lds_byte), "v"(gaddr)
                 : "memory");
}
#define WAIT_ASYNCCNT(n) asm volatile("s_wait_asynccnt " #n ::: "memory")

__device__ __forceinline__ unsigned lds_addr_of(const void* p) {
    return (unsigned)(unsigned long long)p;
}

// ---------------------------------------------------------------------------
// fp32 -> bf16 elementwise cast (grid-strided)
// ---------------------------------------------------------------------------
__global__ void cast_f32_bf16_kernel(const float* __restrict__ src,
                                     bf16_t* __restrict__ dst, int n) {
    int i = blockIdx.x * blockDim.x + threadIdx.x;
    int stride = gridDim.x * blockDim.x;
    for (; i < n; i += stride) dst[i] = (bf16_t)src[i];
}

// ---------------------------------------------------------------------------
// GEMM: C[m,n] = sum_k A[m,k] * W[n,k] + bias[n]   (torch Linear semantics)
// A: bf16 [M,K] row-major, W: bf16 [N,K] row-major.
// LAYOUT 0: out fp32 [M,N] row-major       (final projection)
// LAYOUT 1: out bf16 [B,H,T,DK]            (Q, K for attention)
// LAYOUT 2: out bf16 [B,H,DK,T]            (V transposed for P*V B-fragments)
// Block = 256 threads = 8 waves; block tile 64(M) x 32(N); K-step 32.
// A tile double-buffered in LDS via async DMA.
// ---------------------------------------------------------------------------
template <int LAYOUT>
__global__ __launch_bounds__(256) void gemm_wmma_kernel(
    const bf16_t* __restrict__ A, const bf16_t* __restrict__ W,
    const float* __restrict__ bias, void* __restrict__ outp, int Ktot) {

    __shared__ bf16_t ldsA[2][64][40];  // double buffer, padded rows

    const int tid     = threadIdx.x;
    const int w       = tid >> 5;        // wave id 0..7
    const int lane    = tid & 31;
    const int lane_lo = lane & 15;
    const int half    = lane >> 4;
    const int wm      = w & 3;           // M subtile 0..3
    const int wn      = w >> 2;          // N subtile 0..1

    const int m_base = blockIdx.x * 64;
    const int n_base = blockIdx.y * 32;

    // cooperative A-tile staging: 256 threads * 16B = 64x32 bf16 tile
    const int ld_row = tid >> 2;
    const int ld_grp = tid & 3;
    const bf16_t* aRow = A + (size_t)(m_base + ld_row) * Ktot + ld_grp * 8;
    const unsigned myLds[2] = {
        lds_addr_of(&ldsA[0][ld_row][ld_grp * 8]),
        lds_addr_of(&ldsA[1][ld_row][ld_grp * 8])};

    v8f acc = {0.f, 0.f, 0.f, 0.f, 0.f, 0.f, 0.f, 0.f};

    const bf16_t* wRow = W + (size_t)(n_base + wn * 16 + lane_lo) * Ktot;

    const int nch = Ktot >> 5;
    // prologue: start DMA of chunk 0 into buffer 0
    async_copy_b128(myLds[0], (unsigned long long)aRow);

    for (int j = 0; j < nch; ++j) {
        const int k0  = j * 32;
        const int cur = j & 1;
        if (j + 1 < nch) {
            // stream chunk j+1 into the other buffer while we compute
            async_copy_b128(myLds[cur ^ 1],
                            (unsigned long long)(aRow + k0 + 32));
            __builtin_prefetch(aRow + k0 + 64, 0, 0);  // global_prefetch_b8
            WAIT_ASYNCCNT(1);  // in-order: chunk j has landed
        } else {
            WAIT_ASYNCCNT(0);
        }
        __syncthreads();  // chunk j visible to all waves

        // A fragment from LDS (row = tile row lane_lo, per-ISA K striping)
        const bf16_t* lrow = &ldsA[cur][wm * 16 + lane_lo][0];
        v16bf afrag, bfrag;
#pragma unroll
        for (int idx = 0; idx < 16; ++idx) afrag[idx] = lrow[kmap(idx, half)];
        // B fragment: column n = W row (n_base + wn*16 + lane_lo)
        const bf16_t* wp = wRow + k0;
#pragma unroll
        for (int idx = 0; idx < 16; ++idx) bfrag[idx] = wp[kmap(idx, half)];

        acc = __builtin_amdgcn_wmma_f32_16x16x32_bf16(
            false, afrag, false, bfrag, (short)0, acc, false, false);

        __syncthreads();  // all waves done with buffer `cur` before reuse
    }

    const int n  = n_base + wn * 16 + lane_lo;
    const float bvv = bias[n];
#pragma unroll
    for (int r = 0; r < 8; ++r) {
        const int m = m_base + wm * 16 + r + 8 * half;
        const float val = acc[r] + bvv;
        if (LAYOUT == 0) {
            ((float*)outp)[(size_t)m * Dc + n] = val;
        } else {
            const int b = m / Tc, t = m % Tc, h = n / DKc, dk = n % DKc;
            if (LAYOUT == 1)
                ((bf16_t*)outp)[(((size_t)(b * Hc + h) * Tc) + t) * DKc + dk] = (bf16_t)val;
            else
                ((bf16_t*)outp)[(((size_t)(b * Hc + h) * DKc) + dk) * Tc + t] = (bf16_t)val;
        }
    }
}

// ---------------------------------------------------------------------------
// Causal flash attention: one wave per (b, h, 16-row query tile).
// Q,K in [B,H,T,DK] bf16; V in [B,H,DK,T] bf16. Output ctx bf16 [B*T, D].
// Online softmax state lives per-lane: VGPR r holds row (r + 8*half).
// ---------------------------------------------------------------------------
__global__ __launch_bounds__(128) void attn_wmma_kernel(
    const bf16_t* __restrict__ q_ws, const bf16_t* __restrict__ k_ws,
    const bf16_t* __restrict__ vt_ws, bf16_t* __restrict__ ctx) {

    __shared__ float sP[4][16][33];  // per-wave P relayout staging (padded)

    const int w       = threadIdx.x >> 5;
    const int lane    = threadIdx.x & 31;
    const int lane_lo = lane & 15;
    const int half    = lane >> 4;

    const int id    = blockIdx.x * 4 + w;
    const int qtile = id & 127;
    const int h     = (id >> 7) & 15;
    const int b     = id >> 11;
    const int q0    = qtile * 16;

    const size_t bhT = (size_t)(b * Hc + h) * Tc;

    // Q A-fragments (16x64 = two 16x32 chunks), loaded once
    const bf16_t* qbase = q_ws + (bhT + q0 + lane_lo) * DKc;
    v16bf qA[2];
#pragma unroll
    for (int c = 0; c < 2; ++c)
#pragma unroll
        for (int idx = 0; idx < 16; ++idx)
            qA[c][idx] = qbase[c * 32 + kmap(idx, half)];

    v8f o[4];
#pragma unroll
    for (int t = 0; t < 4; ++t)
#pragma unroll
        for (int r = 0; r < 8; ++r) o[t][r] = 0.f;

    float m_run[8], l_run[8];
#pragma unroll
    for (int r = 0; r < 8; ++r) { m_run[r] = -3.0e38f; l_run[r] = 0.f; }

    const float scale = 0.125f;  // 1/sqrt(64)
    const int nchunks = (q0 + 16 + 31) >> 5;  // causal early exit

    for (int j = 0; j < nchunks; ++j) {
        const int kv0 = j * 32;
        // prefetch next KV chunk while this one computes
        if (j + 1 < nchunks) {
            __builtin_prefetch(k_ws + (bhT + kv0 + 32 + lane_lo) * DKc, 0, 0);
            __builtin_prefetch(
                vt_ws + ((size_t)(b * Hc + h) * DKc + lane_lo) * Tc + kv0 + 32, 0, 0);
        }

        // ---- scores S (16x32) = Q(16x64) . K^T, two 16-key subtiles ----
        v8f s[2];
#pragma unroll
        for (int sub = 0; sub < 2; ++sub) {
            const int k0s = kv0 + sub * 16;
            const bf16_t* kbase = k_ws + (bhT + k0s + lane_lo) * DKc;
            v16bf kB0, kB1;
#pragma unroll
            for (int idx = 0; idx < 16; ++idx) {
                kB0[idx] = kbase[kmap(idx, half)];
                kB1[idx] = kbase[32 + kmap(idx, half)];
            }
            v8f a = {0.f, 0.f, 0.f, 0.f, 0.f, 0.f, 0.f, 0.f};
            a = __builtin_amdgcn_wmma_f32_16x16x32_bf16(false, qA[0], false, kB0,
                                                        (short)0, a, false, false);
            a = __builtin_amdgcn_wmma_f32_16x16x32_bf16(false, qA[1], false, kB1,
                                                        (short)0, a, false, false);
            const int ng = k0s + lane_lo;  // key index for this lane's column
#pragma unroll
            for (int r = 0; r < 8; ++r) {
                const int mg = q0 + r + 8 * half;
                s[sub][r] = (ng > mg) ? -1e9f : a[r] * scale;
            }
        }

        // ---- online softmax: row reductions across the 16-lane group ----
        float mx[8];
#pragma unroll
        for (int r = 0; r < 8; ++r) mx[r] = fmaxf(s[0][r], s[1][r]);
#pragma unroll
        for (int msk = 1; msk < 16; msk <<= 1)
#pragma unroll
            for (int r = 0; r < 8; ++r)
                mx[r] = fmaxf(mx[r], __shfl_xor(mx[r], msk, 32));

        float al[8], rowsum[8];
#pragma unroll
        for (int r = 0; r < 8; ++r) {
            const float mn = fmaxf(m_run[r], mx[r]);
            al[r] = __expf(m_run[r] - mn);
            const float p0 = __expf(s[0][r] - mn);
            const float p1 = __expf(s[1][r] - mn);
            s[0][r] = p0; s[1][r] = p1;
            rowsum[r] = p0 + p1;
            m_run[r] = mn;
        }
#pragma unroll
        for (int msk = 1; msk < 16; msk <<= 1)
#pragma unroll
            for (int r = 0; r < 8; ++r)
                rowsum[r] += __shfl_xor(rowsum[r], msk, 32);
#pragma unroll
        for (int r = 0; r < 8; ++r) l_run[r] = l_run[r] * al[r] + rowsum[r];
#pragma unroll
        for (int t = 0; t < 4; ++t)
#pragma unroll
            for (int r = 0; r < 8; ++r) o[t][r] *= al[r];

        // ---- relayout P: C-layout f32 -> A-layout bf16 via LDS ----
#pragma unroll
        for (int r = 0; r < 8; ++r) {
            sP[w][r + 8 * half][lane_lo]      = s[0][r];
            sP[w][r + 8 * half][16 + lane_lo] = s[1][r];
        }
        asm volatile("s_wait_dscnt 0" ::: "memory");  // in-wave cross-lane LDS
        v16bf pA;
#pragma unroll
        for (int idx = 0; idx < 16; ++idx)
            pA[idx] = (bf16_t)sP[w][lane_lo][kmap(idx, half)];

        // ---- O(16x64) += P(16x32) . V(32x64), V^T layout -> contiguous ----
#pragma unroll
        for (int t = 0; t < 4; ++t) {
            const bf16_t* vcol =
                vt_ws + ((size_t)(b * Hc + h) * DKc + t * 16 + lane_lo) * Tc + kv0;
            v16bf vB;
#pragma unroll
            for (int idx = 0; idx < 16; ++idx) vB[idx] = vcol[kmap(idx, half)];
            o[t] = __builtin_amdgcn_wmma_f32_16x16x32_bf16(
                false, pA, false, vB, (short)0, o[t], false, false);
        }
    }

    // ---- epilogue: normalize and emit ctx bf16 row-major [B*T, D] ----
    float inv[8];
#pragma unroll
    for (int r = 0; r < 8; ++r) inv[r] = 1.f / l_run[r];
#pragma unroll
    for (int t = 0; t < 4; ++t)
#pragma unroll
        for (int r = 0; r < 8; ++r) {
            const int m = q0 + r + 8 * half;
            const int d = h * DKc + t * 16 + lane_lo;
            ctx[((size_t)(b * Tc + m)) * Dc + d] = (bf16_t)(o[t][r] * inv[r]);
        }
}

// ---------------------------------------------------------------------------
// launcher
// ---------------------------------------------------------------------------
extern "C" void kernel_launch(void* const* d_in, const int* in_sizes, int n_in,
                              void* d_out, int out_size, void* d_ws, size_t ws_size,
                              hipStream_t stream) {
    const float* x  = (const float*)d_in[0];
    const float* Wq = (const float*)d_in[1];
    const float* bq = (const float*)d_in[2];
    const float* Wk = (const float*)d_in[3];
    const float* bk = (const float*)d_in[4];
    const float* Wv = (const float*)d_in[5];
    const float* bv = (const float*)d_in[6];
    const float* Wo = (const float*)d_in[7];
    const float* bo = (const float*)d_in[8];
    float* out = (float*)d_out;

    char* ws = (char*)d_ws;
    const size_t szX = (size_t)BTc * Dc * sizeof(bf16_t);   // 16 MB
    const size_t szW = (size_t)Dc * Dc * sizeof(bf16_t);    // 2 MB
    bf16_t* xb   = (bf16_t*)(ws);
    bf16_t* wqb  = (bf16_t*)(ws + szX);
    bf16_t* wkb  = (bf16_t*)(ws + szX + szW);
    bf16_t* wvb  = (bf16_t*)(ws + szX + 2 * szW);
    bf16_t* wob  = (bf16_t*)(ws + szX + 3 * szW);
    bf16_t* q_ws = (bf16_t*)(ws + szX + 4 * szW);
    bf16_t* k_ws = (bf16_t*)(ws + 2 * szX + 4 * szW);
    bf16_t* vt_ws= (bf16_t*)(ws + 3 * szX + 4 * szW);
    bf16_t* ctx  = (bf16_t*)(ws + 4 * szX + 4 * szW);

    // 1) casts
    cast_f32_bf16_kernel<<<2048, 256, 0, stream>>>(x,  xb,  BTc * Dc);
    cast_f32_bf16_kernel<<<1024, 256, 0, stream>>>(Wq, wqb, Dc * Dc);
    cast_f32_bf16_kernel<<<1024, 256, 0, stream>>>(Wk, wkb, Dc * Dc);
    cast_f32_bf16_kernel<<<1024, 256, 0, stream>>>(Wv, wvb, Dc * Dc);
    cast_f32_bf16_kernel<<<1024, 256, 0, stream>>>(Wo, wob, Dc * Dc);

    // 2) QKV projections (WMMA GEMMs, async-LDS double-buffered)
    dim3 ggrid(BTc / 64, Dc / 32);
    gemm_wmma_kernel<1><<<ggrid, 256, 0, stream>>>(xb, wqb, bq, q_ws,  Dc);
    gemm_wmma_kernel<1><<<ggrid, 256, 0, stream>>>(xb, wkb, bk, k_ws,  Dc);
    gemm_wmma_kernel<2><<<ggrid, 256, 0, stream>>>(xb, wvb, bv, vt_ws, Dc);

    // 3) causal flash attention (WMMA)
    attn_wmma_kernel<<<(Bc * Hc * (Tc / 16)) / 4, 128, 0, stream>>>(q_ws, k_ws, vt_ws, ctx);

    // 4) output projection (WMMA, fp32 out)
    gemm_wmma_kernel<0><<<ggrid, 256, 0, stream>>>(ctx, wob, bo, out, Dc);
}
